// IDS4Block_55516747268648
// MI455X (gfx1250) — matrix operand order repaired and checked by reading, not compiled, not run on hardware
//
#include <hip/hip_runtime.h>
#include <hip/hip_bf16.h>
#include <math.h>
#include <stdint.h>

// Problem sizes (fixed by reference)
#define Bx 4
#define Lx 2048
#define DMx 1024
#define DSx 64
#define SSx 4096      // DS*DS
#define BLx 8192      // B*L
#define Gx 64         // scan chunk length
#define NCx 32        // chunks per batch (L/G)

#define USE_ASYNC_LDS 1

#define SA 40         // bf16 LDS stride: 80B rows -> every 8-elem group 16B aligned
#define SF 66         // f32 LDS stride: even -> v2f loads 8B aligned

typedef __attribute__((ext_vector_type(16))) __bf16 v16bf;
typedef __attribute__((ext_vector_type(8)))  __bf16 v8bf;
typedef __attribute__((ext_vector_type(4)))  __bf16 v4bf;
typedef __attribute__((ext_vector_type(8)))  float  v8f;
typedef __attribute__((ext_vector_type(4)))  float  v4f;
typedef __attribute__((ext_vector_type(2)))  float  v2f;

// ---------------- bf16 WMMA fragment builders (16x16x32, f32 accum) ----------
// A tile row-major (stride SA). ISA 7.12.2: lanes 0-15 row M, elems 0-7 ->
// K=0..7, 8-15 -> K=16..23; lanes 16-31: K=8..15 / K=24..31. Each 8-elem group
// is 16 contiguous 16B-aligned bytes -> ds_load_b128.
__device__ inline v16bf load_afrag_bf16(const __bf16* As, int mo) {
    int l  = threadIdx.x & 31;
    int m  = mo + (l & 15);
    int k0 = (l < 16) ? 0  : 8;
    int k1 = (l < 16) ? 16 : 24;
    const __bf16* row = As + m * SA;
    v8bf lo = *(const v8bf*)(row + k0);
    v8bf hi = *(const v8bf*)(row + k1);
    return __builtin_shufflevector(lo, hi, 0, 1, 2, 3, 4, 5, 6, 7,
                                   8, 9, 10, 11, 12, 13, 14, 15);
}
// B tile stored K-transposed: BsT[n*SA + k]. Lane n=l&15 (+co), K=0..7/16..23 for
// lanes 0-15, 8..15/24..31 for lanes 16-31 (matches A) -> 2x ds_load_b128.
__device__ inline v16bf load_bfrag_bf16(const __bf16* BsT, int co) {
    int l  = threadIdx.x & 31;
    int n  = co + (l & 15);
    int kb = (l < 16) ? 0 : 8;
    const __bf16* row = BsT + n * SA + kb;
    v8bf lo = *(const v8bf*)(row);
    v8bf hi = *(const v8bf*)(row + 16);
    return __builtin_shufflevector(lo, hi, 0, 1, 2, 3, 4, 5, 6, 7,
                                   8, 9, 10, 11, 12, 13, 14, 15);
}

// ---------------- async global->LDS transposed copy (CDNA5 path) -------------
// dstT[c*SF + r] = src[r*64 + c]; 4096 f32, 256 threads. Per-lane independent
// global/LDS addresses let the async engine do the transpose.
__device__ inline void async_copy_T64(const float* __restrict__ src, float* dstT) {
#pragma unroll
    for (int i = 0; i < 16; ++i) {
        int e = threadIdx.x + i * 256;
        int r = e >> 6, c = e & 63;
#if USE_ASYNC_LDS
        unsigned lds          = (unsigned)(uintptr_t)(dstT + c * SF + r);
        unsigned long long ga = (unsigned long long)(uintptr_t)(src + e);
        asm volatile("global_load_async_to_lds_b32 %0, %1, off"
                     :: "v"(lds), "v"(ga) : "memory");
#else
        dstT[c * SF + r] = src[e];
#endif
    }
}
__device__ inline void async_wait_all() {
#if USE_ASYNC_LDS
    asm volatile("s_wait_asynccnt 0x0" ::: "memory");
#endif
}

// ---------------- 64x64 f32 matmul via v_wmma_f32_16x16x4_f32 -----------------
// Pn = Pc @ M; Pc row-major (stride SF), M K-transposed (MbT[n*SF+k]).
// 256 threads = 8 waves x 2 tiles; operands are 8B-aligned v2f (ds_load_b64).
__device__ inline void matmul64(const float* Pc, const float* MbT, float* Pn,
                                float* gout) {
    int w    = threadIdx.x >> 5;
    int l    = threadIdx.x & 31;
    int half = (l < 16) ? 0 : 2;
    int ln   = l & 15;
#pragma unroll
    for (int h = 0; h < 2; ++h) {
        int ti = w + h * 8;           // 0..15
        int mo = (ti >> 2) * 16;
        int no = (ti & 3) * 16;
        const float* ap = Pc  + (mo + ln) * SF + half;
        const float* bp = MbT + (no + ln) * SF + half;
        v8f acc = {};
#pragma unroll
        for (int kk = 0; kk < 16; ++kk) {
            v2f a = *(const v2f*)(ap + kk * 4);
            v2f b = *(const v2f*)(bp + kk * 4);
            acc = __builtin_amdgcn_wmma_f32_16x16x4_f32(
                false, a, false, b, (short)0, acc, false, false);
        }
        int n  = no + ln;
        int mb = mo + ((l < 16) ? 0 : 8);
#pragma unroll
        for (int r = 0; r < 8; ++r) {
            if (Pn)   Pn[(mb + r) * SF + n]           = acc[r];
            if (gout) gout[(size_t)(mb + r) * 64 + n] = acc[r];
        }
    }
}

// ============ Kernel 0a: f32 -> bf16 elementwise convert ======================
__global__ __launch_bounds__(256) void k0_cvt(const float* __restrict__ src,
                                              __bf16* __restrict__ dst, int n4) {
    int i = blockIdx.x * 256 + threadIdx.x;
    if (i >= n4) return;
    v4f v = ((const v4f*)src)[i];
    v4bf o;
    o[0] = (__bf16)v[0]; o[1] = (__bf16)v[1];
    o[2] = (__bf16)v[2]; o[3] = (__bf16)v[3];
    ((v4bf*)dst)[i] = o;
}

// ============ Kernel 0b: A (k-major f32) -> WT[n][k] bf16 (one-time) ==========
__global__ __launch_bounds__(256) void k0_transposeA(const float* __restrict__ A,
                                                     __bf16* __restrict__ WT) {
    __shared__ __bf16 tile[64 * 65];
    int kb = blockIdx.x * 64;
    int nb = blockIdx.y * 64;
#pragma unroll
    for (int i = 0; i < 16; ++i) {
        int e = threadIdx.x + i * 256;
        int k = e >> 6, n = e & 63;                     // coalesced in n
        tile[n * 65 + k] = (__bf16)A[(size_t)(kb + k) * SSx + nb + n];
    }
    __syncthreads();
#pragma unroll
    for (int i = 0; i < 16; ++i) {
        int e = threadIdx.x + i * 256;
        int n = e >> 6, k = e & 63;                     // coalesced in k
        WT[(size_t)(nb + n) * DMx + kb + k] = tile[n * 65 + k];
    }
}

// ============ Kernel 1: Ax = gelu(xbf @ WT^T), bf16 WMMA ======================
// M=8192, K=1024, N=4096. 256 thr -> 32x128 tile; 2 WMMA/wave/step; LDS
// double-buffered, single barrier per K-step. All staging = aligned b64/b128 copies.
__device__ inline void k1_stage(const __bf16* __restrict__ xbf,
                                const __bf16* __restrict__ WT,
                                __bf16* Xs, __bf16* Ws, int m0, int n0, int kk) {
    {   // X tile 32x32: 4 contiguous bf16 per thread (b64)
        int e = threadIdx.x * 4;
        int r = e >> 5, c = e & 31;
        *(uint64_t*)(Xs + r * SA + c) =
            *(const uint64_t*)(xbf + (size_t)(m0 + r) * DMx + kk * 32 + c);
    }
    {   // W tile 128n x 32k from WT[n][k]: 16 contiguous bf16 per thread (2x b128)
        int n  = threadIdx.x >> 1;
        int kb = (threadIdx.x & 1) * 16;
        const __bf16* g = WT + (size_t)(n0 + n) * DMx + kk * 32 + kb;
        *(v8bf*)(Ws + n * SA + kb)     = *(const v8bf*)(g);
        *(v8bf*)(Ws + n * SA + kb + 8) = *(const v8bf*)(g + 8);
    }
}

__global__ __launch_bounds__(256) void k1_gemm_gelu(const __bf16* __restrict__ xbf,
                                                    const __bf16* __restrict__ WT,
                                                    float* __restrict__ Ax) {
    __shared__ __attribute__((aligned(16))) __bf16 Xs[2][32 * SA];
    __shared__ __attribute__((aligned(16))) __bf16 Ws[2][128 * SA];
    int m0 = blockIdx.x * 32;
    int n0 = blockIdx.y * 128;
    int w  = threadIdx.x >> 5;
    int mo = (w >> 2) * 16;
    int no = (w & 3) * 16;
    v8f acc0 = {}, acc1 = {};
    k1_stage(xbf, WT, Xs[0], Ws[0], m0, n0, 0);
    __syncthreads();
    for (int kk = 0; kk < 32; ++kk) {
        int cur = kk & 1;
        if (kk + 1 < 32) k1_stage(xbf, WT, Xs[cur ^ 1], Ws[cur ^ 1], m0, n0, kk + 1);
        if (kk + 2 < 32)
            __builtin_prefetch(WT + (size_t)(n0 + (threadIdx.x >> 1)) * DMx + (kk + 2) * 32, 0, 3);
        v16bf a  = load_afrag_bf16(Xs[cur], mo);
        v16bf b0 = load_bfrag_bf16(Ws[cur], no);
        v16bf b1 = load_bfrag_bf16(Ws[cur] + 64 * SA, no);
        acc0 = __builtin_amdgcn_wmma_f32_16x16x32_bf16(
            false, a, false, b0, (short)0, acc0, false, false);
        acc1 = __builtin_amdgcn_wmma_f32_16x16x32_bf16(
            false, a, false, b1, (short)0, acc1, false, false);
        __syncthreads();
    }
    int l  = threadIdx.x & 31;
    int n  = n0 + no + (l & 15);
    int mb = m0 + mo + ((l < 16) ? 0 : 8);
#pragma unroll
    for (int r = 0; r < 8; ++r) {
        float v0 = acc0[r];
        float v1 = acc1[r];
        v0 = 0.5f * v0 * (1.0f + erff(v0 * 0.70710678118654752f));
        v1 = 0.5f * v1 * (1.0f + erff(v1 * 0.70710678118654752f));
        Ax[(size_t)(mb + r) * SSx + n]      = v0;
        Ax[(size_t)(mb + r) * SSx + n + 64] = v1;
    }
}

// ============ Kernel 2: per-chunk inclusive scan (f32 WMMA) ===================
__global__ __launch_bounds__(256) void k2_chunkscan(float* __restrict__ Ax,
                                                    float* __restrict__ cprod) {
    __shared__ float P0[64 * SF];
    __shared__ float P1[64 * SF];
    __shared__ float Mb0[64 * SF];
    __shared__ float Mb1[64 * SF];
    int blk  = blockIdx.x;
    int b    = blk >> 5;
    int c    = blk & 31;
    int base = b * Lx + c * Gx;
    for (int e = threadIdx.x; e < SSx; e += 256)        // cum[0] = M0 (row-major)
        P0[(e >> 6) * SF + (e & 63)] = Ax[(size_t)base * SSx + e];
    async_copy_T64(Ax + (size_t)(base + 1) * SSx, Mb0); // preload M1 (transposed)
    async_wait_all();
    __syncthreads();
    float* Pc = P0;
    float* Pn = P1;
    for (int t = 1; t < Gx; ++t) {
        float* cur = (t & 1) ? Mb0 : Mb1;
        float* nxt = (t & 1) ? Mb1 : Mb0;
        if (t + 1 < Gx)                                  // overlap load of M_{t+1}
            async_copy_T64(Ax + (size_t)(base + t + 1) * SSx, nxt);
        matmul64(Pc, cur, Pn, Ax + (size_t)(base + t) * SSx);   // cum[t]
        async_wait_all();
        __syncthreads();
        float* tmp = Pc; Pc = Pn; Pn = tmp;
    }
    for (int e = threadIdx.x; e < SSx; e += 256)
        cprod[(size_t)blk * SSx + e] = Pc[(e >> 6) * SF + (e & 63)];
}

// ============ Kernel 3: exclusive prefix of chunk products (per batch) ========
__global__ __launch_bounds__(256) void k3_chunkpref(const float* __restrict__ cprod,
                                                    float* __restrict__ cpref) {
    __shared__ float P0[64 * SF];
    __shared__ float P1[64 * SF];
    __shared__ float Mb0[64 * SF];
    __shared__ float Mb1[64 * SF];
    int b = blockIdx.x;
    for (int e = threadIdx.x; e < SSx; e += 256) {
        int r = e >> 6, cc = e & 63;
        float v = (r == cc) ? 1.0f : 0.0f;
        P0[r * SF + cc] = v;
        cpref[(size_t)(b * NCx) * SSx + e] = v;          // prefix[0] = I
    }
    async_copy_T64(cprod + (size_t)(b * NCx) * SSx, Mb0);
    async_wait_all();
    __syncthreads();
    float* Pc = P0;
    float* Pn = P1;
    for (int c = 1; c < NCx; ++c) {
        float* cur = (c & 1) ? Mb0 : Mb1;
        float* nxt = (c & 1) ? Mb1 : Mb0;
        if (c + 1 < NCx)
            async_copy_T64(cprod + (size_t)(b * NCx + c) * SSx, nxt);
        matmul64(Pc, cur, Pn, cpref + (size_t)(b * NCx + c) * SSx);
        async_wait_all();
        __syncthreads();
        float* tmp = Pc; Pc = Pn; Pn = tmp;
    }
}

// ============ Kernel 4: global = prefix @ local, fused proj (bf16 p out) ======
__global__ __launch_bounds__(256) void k4_apply_proj(const float* __restrict__ cum,
                                                     const float* __restrict__ cpref,
                                                     const float* __restrict__ proj_w,
                                                     const float* __restrict__ proj_b,
                                                     __bf16* __restrict__ p) {
    __shared__ float Pref[64 * SF];
    __shared__ float Gb[64 * SF];
    __shared__ float Mb0[64 * SF];
    __shared__ float Mb1[64 * SF];
    __shared__ float pw[64];
    int blk  = blockIdx.x;
    int b    = blk >> 5;
    int c    = blk & 31;
    int base = b * Lx + c * Gx;
    for (int e = threadIdx.x; e < SSx; e += 256)
        Pref[(e >> 6) * SF + (e & 63)] = cpref[(size_t)blk * SSx + e];
    if (threadIdx.x < 64) pw[threadIdx.x] = proj_w[threadIdx.x];
    async_copy_T64(cum + (size_t)base * SSx, Mb0);       // preload local cum[0]
    async_wait_all();
    __syncthreads();
    float pb = proj_b[0];
    for (int t = 0; t < Gx; ++t) {
        float* cur = (t & 1) ? Mb1 : Mb0;
        float* nxt = (t & 1) ? Mb0 : Mb1;
        if (t + 1 < Gx)
            async_copy_T64(cum + (size_t)(base + t + 1) * SSx, nxt);
        matmul64(Pref, cur, Gb, nullptr);                // G = prefix @ local
        async_wait_all();
        __syncthreads();
        if (threadIdx.x < 64) {
            int s = threadIdx.x;
            float acc = 0.0f;
#pragma unroll
            for (int e2 = 0; e2 < 64; ++e2) acc += Gb[s * SF + e2] * pw[e2];
            p[(size_t)(base + t) * DSx + s] = (__bf16)(acc + pb);
        }
        __syncthreads();                                 // Gb reused next iter
    }
}

// ============ Kernel 5: out = p@C_w^T + x@D_w^T + C_b + D_b (bf16 WMMA) =======
// M=8192, N=1024, K = 64 + 1024. 256 thr, 32x64 tile, double-buffered LDS.
__device__ inline void k5_stage(const __bf16* __restrict__ xbf,
                                const __bf16* __restrict__ pbf,
                                const __bf16* __restrict__ Cw,
                                const __bf16* __restrict__ Dw,
                                __bf16* As, __bf16* Bs, int m0, int n0, int s) {
    {   // A tile 32x32: 4 contiguous bf16 per thread (b64)
        int e = threadIdx.x * 4;
        int r = e >> 5, c = e & 31;
        const __bf16* g = (s < 2)
            ? (pbf + (size_t)(m0 + r) * DSx + s * 32 + c)
            : (xbf + (size_t)(m0 + r) * DMx + (s - 2) * 32 + c);
        *(uint64_t*)(As + r * SA + c) = *(const uint64_t*)g;
    }
    {   // B tile 64n x 32k: 8 contiguous bf16 per thread (b128)
        int n  = threadIdx.x >> 2;
        int kb = (threadIdx.x & 3) * 8;
        const __bf16* g = (s < 2)
            ? (Cw + (size_t)(n0 + n) * DSx + s * 32 + kb)
            : (Dw + (size_t)(n0 + n) * DMx + (s - 2) * 32 + kb);
        *(v8bf*)(Bs + n * SA + kb) = *(const v8bf*)g;
    }
}

__global__ __launch_bounds__(256) void k5_out(const __bf16* __restrict__ xbf,
                                              const __bf16* __restrict__ pbf,
                                              const __bf16* __restrict__ Cw,
                                              const float* __restrict__ C_b,
                                              const __bf16* __restrict__ Dw,
                                              const float* __restrict__ D_b,
                                              float* __restrict__ out) {
    __shared__ __attribute__((aligned(16))) __bf16 As[2][32 * SA];
    __shared__ __attribute__((aligned(16))) __bf16 Bs[2][64 * SA];
    int m0 = blockIdx.x * 32;
    int n0 = blockIdx.y * 64;
    int w  = threadIdx.x >> 5;
    int mo = (w >> 2) * 16;
    int no = (w & 3) * 16;
    v8f acc = {};
    k5_stage(xbf, pbf, Cw, Dw, As[0], Bs[0], m0, n0, 0);
    __syncthreads();
    for (int s = 0; s < 34; ++s) {
        int cur = s & 1;
        if (s + 1 < 34)
            k5_stage(xbf, pbf, Cw, Dw, As[cur ^ 1], Bs[cur ^ 1], m0, n0, s + 1);
        v16bf a = load_afrag_bf16(As[cur], mo);
        v16bf b = load_bfrag_bf16(Bs[cur], no);
        acc = __builtin_amdgcn_wmma_f32_16x16x32_bf16(
            false, a, false, b, (short)0, acc, false, false);
        __syncthreads();
    }
    int l  = threadIdx.x & 31;
    int n  = n0 + no + (l & 15);
    int mb = m0 + mo + ((l < 16) ? 0 : 8);
    float bias = C_b[n] + D_b[n];
#pragma unroll
    for (int r = 0; r < 8; ++r)
        out[(size_t)(mb + r) * DMx + n] = acc[r] + bias;
}

// ============================ launcher =======================================
extern "C" void kernel_launch(void* const* d_in, const int* in_sizes, int n_in,
                              void* d_out, int out_size, void* d_ws, size_t ws_size,
                              hipStream_t stream) {
    const float* x      = (const float*)d_in[0];
    const float* A      = (const float*)d_in[1];
    const float* proj_w = (const float*)d_in[2];
    const float* proj_b = (const float*)d_in[3];
    const float* C_w    = (const float*)d_in[4];
    const float* C_b    = (const float*)d_in[5];
    const float* D_w    = (const float*)d_in[6];
    const float* D_b    = (const float*)d_in[7];
    float* out = (float*)d_out;

    // Workspace: f32 region (Ax/cum in-place, chunk products, prefixes) then
    // bf16 region (pre-converted operands + p).
    float*  ws    = (float*)d_ws;
    float*  Ax    = ws;                                  // BLx*SSx f32
    float*  cprod = Ax    + (size_t)BLx * SSx;           // Bx*NCx*SSx
    float*  cpref = cprod + (size_t)Bx * NCx * SSx;
    __bf16* xbf   = (__bf16*)(cpref + (size_t)Bx * NCx * SSx);
    __bf16* WT    = xbf  + (size_t)BLx * DMx;            // A^T bf16 [n][k]
    __bf16* Dwbf  = WT   + (size_t)SSx * DMx;
    __bf16* Cwbf  = Dwbf + (size_t)DMx * DMx;
    __bf16* pbf   = Cwbf + (size_t)DMx * DSx;            // BLx*DSx bf16

    // One-time precision/layout prep (all L2-resident afterwards).
    k0_cvt<<<(BLx * DMx / 4 + 255) / 256, 256, 0, stream>>>(x,   xbf,  BLx * DMx / 4);
    k0_cvt<<<(DMx * DMx / 4 + 255) / 256, 256, 0, stream>>>(D_w, Dwbf, DMx * DMx / 4);
    k0_cvt<<<(DMx * DSx / 4 + 255) / 256, 256, 0, stream>>>(C_w, Cwbf, DMx * DSx / 4);
    k0_transposeA<<<dim3(DMx / 64, SSx / 64), 256, 0, stream>>>(A, WT);

    k1_gemm_gelu<<<dim3(BLx / 32, SSx / 128), 256, 0, stream>>>(xbf, WT, Ax);
    k2_chunkscan<<<Bx * NCx, 256, 0, stream>>>(Ax, cprod);
    k3_chunkpref<<<Bx, 256, 0, stream>>>(cprod, cpref);
    k4_apply_proj<<<Bx * NCx, 256, 0, stream>>>(Ax, cpref, proj_w, proj_b, pbf);
    k5_out<<<dim3(BLx / 32, DMx / 64), 256, 0, stream>>>(xbf, pbf, Cwbf, C_b, Dwbf, D_b, out);
}
